// Dance_9594956939781
// MI455X (gfx1250) — matrix operand order repaired and checked by reading, not compiled, not run on hardware
//
#include <hip/hip_runtime.h>
#include <math.h>

// ---------------------------------------------------------------------------
// CDNA5 (gfx1250) wave32 WMMA types & helpers
// ---------------------------------------------------------------------------
typedef __attribute__((ext_vector_type(16))) __bf16 v16bf;
typedef __attribute__((ext_vector_type(2)))  __bf16 v2bf;
typedef __attribute__((ext_vector_type(8)))  float  v8f;

union FragU { float4 q[2]; v16bf v; };

#define WMMA_BF16(a, b, c) \
  __builtin_amdgcn_wmma_f32_16x16x32_bf16(false, (a), false, (b), (short)0, (c), false, false)

#define LD128(p) (*reinterpret_cast<const float4*>(p))

static __device__ __forceinline__ float sigf(float x) { return 1.f / (1.f + __expf(-x)); }

// Fragment index math (CDNA5 ISA 7.12.2, 16-bit 16x16x32):
//  A (16xK chunk): lane L -> row M = L&15 ; elements 0..7 hold K contiguous at
//                  kc+8*(L>>4), elements 8..15 at kc+16+8*(L>>4)  -> 2x b128
//  B (Kx16):       lane L -> col N = L&15 ; same K split as A: kc+8*(L>>4), kc+16+8*(L>>4)
//  C/D f32:        lane L -> col N = L&15 ; element v -> row M = v + 8*(L>>4)
// Convs are split per-tap so the reduction axis is pure channel (contiguous in
// a transposed LDS slab); every WMMA is fed by exactly 4x ds_load_b128.

// ---------------------------------------------------------------------------
// Implicit-GEMM 3x3 conv (pad=1), NCHW fp32 in/out, bf16 WMMA, f32 accumulate.
// Block = 8 waves = 128 consecutive output pixels of one row x 16 out channels.
// Per 32-channel chunk: stage transposed halo slab slabT3[3*130 rows][40 c]
// and per-tap weights wst[9][16][40]; 9 taps x 1 wmma per chunk.
// ---------------------------------------------------------------------------
__global__ __launch_bounds__(256) void k_conv3x3_wmma(
    const float* __restrict__ x, const float* __restrict__ w, float* __restrict__ y,
    int Bn, int Cin, int H, int W, int Cout, int yC, int yc0) {
  extern __shared__ char smem[];
  __bf16* slabT = (__bf16*)smem;                    // [390 rows][40 c] (32 used)
  __bf16* wst   = (__bf16*)(smem + 390 * 40 * 2);   // [9t][16n][40 c]
  const int K = Cin * 9;
  const int tid = threadIdx.x;
  const int oc0 = blockIdx.y << 4;
  const int prow = (blockIdx.x * 128) / W;
  const int oxb  = (blockIdx.x * 128) % W;
  const int oy   = prow % H;
  const int b    = prow / H;
  const int wave = tid >> 5, lane = tid & 31;
  const int half = lane >> 4, lm = lane & 15;
  v8f acc = {0.f, 0.f, 0.f, 0.f, 0.f, 0.f, 0.f, 0.f};
  const int nchunk = (Cin + 31) >> 5;
  for (int ci = 0; ci < nchunk; ++ci) {
    const int c0 = ci << 5;
    __syncthreads();
    // stage transposed input slab: rows = (r,col), cols = channel pair
    for (int idx = tid; idx < 16 * 390; idx += 256) {
      int cpair = idx / 390;
      int row   = idx - cpair * 390;
      int r     = row / 130;
      int col   = row - r * 130;
      int iy = oy + r - 1;
      int ix = oxb + col - 1;
      int c  = c0 + (cpair << 1);
      bool okp = (ix >= 0 && ix < W && iy >= 0 && iy < H);
      const float* xs = x + ((size_t)(b * Cin + c) * H + iy) * W + ix;
      v2bf pk;
      pk.x = (__bf16)((okp && c     < Cin) ? xs[0] : 0.f);
      pk.y = (__bf16)((okp && c + 1 < Cin) ? xs[(size_t)H * W] : 0.f);
      *reinterpret_cast<v2bf*>(&slabT[row * 40 + (cpair << 1)]) = pk;
    }
    // stage per-tap transposed weights
    for (int idx = tid; idx < 9 * 16 * 32; idx += 256) {
      int t   = idx >> 9;
      int rem = idx & 511;
      int nn  = rem >> 5;
      int cl  = rem & 31;
      int c   = c0 + cl;
      float f = (c < Cin) ? w[(size_t)(oc0 + nn) * K + c * 9 + t] : 0.f;
      wst[(t * 16 + nn) * 40 + cl] = (__bf16)f;
    }
    __syncthreads();
#pragma unroll
    for (int t = 0; t < 9; ++t) {
      const int ky = t / 3, kx = t - ky * 3;
      FragU A, B;
      const __bf16* ap = slabT + (ky * 130 + (wave << 4) + lm + kx) * 40 + (half << 3);
      A.q[0] = LD128(ap);
      A.q[1] = LD128(ap + 16);
      const __bf16* bp = wst + (t * 16 + lm) * 40 + (half << 3);
      B.q[0] = LD128(bp);
      B.q[1] = LD128(bp + 16);
      acc = WMMA_BF16(A.v, B.v, acc);
    }
  }
  const int oc = oc0 + lm;
  float* yb = y + ((size_t)(b * yC + yc0 + oc) * H + oy) * W + oxb + (wave << 4) + (half << 3);
  *reinterpret_cast<float4*>(yb)     = make_float4(acc[0], acc[1], acc[2], acc[3]);
  *reinterpret_cast<float4*>(yb + 4) = make_float4(acc[4], acc[5], acc[6], acc[7]);
}

// ---------------------------------------------------------------------------
// Circular dilated conv1d as per-tap implicit GEMM. in/out: (P, C, N=128) fp32.
// out[p,o,n] = sum_{c,t} w[o,c,t] * in[p,c,(n+(t-4)*dil) mod N] + bias[o]
// Per 64-channel chunk: slabT[128 n][72 c] + wst[9][16][72]; 9 taps x 2 wmmas.
// ---------------------------------------------------------------------------
__global__ __launch_bounds__(256) void k_circconv_wmma(
    const float* __restrict__ in, const float* __restrict__ w, const float* __restrict__ bias,
    float* __restrict__ out, int P, int Cin, int Cout, int N, int dil) {
  extern __shared__ char smem[];
  __bf16* slabT = (__bf16*)smem;                    // [128 n][72 c] (64 used)
  __bf16* wst   = (__bf16*)(smem + 128 * 72 * 2);   // [9t][16n][72 c]
  const int K = Cin * 9;
  const int tid = threadIdx.x;
  const int oc0 = blockIdx.y << 4;
  const int p   = blockIdx.x;
  const int wave = tid >> 5, lane = tid & 31;
  const int half = lane >> 4, lm = lane & 15;
  const int n0 = wave << 4;
  const float* ip = in + (size_t)p * Cin * N;
  v8f acc = {0.f, 0.f, 0.f, 0.f, 0.f, 0.f, 0.f, 0.f};
  const int nchunk = (Cin + 63) >> 6;
  for (int ci = 0; ci < nchunk; ++ci) {
    const int c0 = ci << 6;
    __syncthreads();
    // stage transposed activations (64 channels)
    for (int idx = tid; idx < 32 * 128; idx += 256) {
      int cpair = idx >> 7;
      int n     = idx & 127;
      int c     = c0 + (cpair << 1);
      v2bf pk;
      pk.x = (__bf16)((c     < Cin) ? ip[(size_t)c * N + n] : 0.f);
      pk.y = (__bf16)((c + 1 < Cin) ? ip[(size_t)(c + 1) * N + n] : 0.f);
      *reinterpret_cast<v2bf*>(&slabT[n * 72 + (cpair << 1)]) = pk;
    }
    // stage per-tap transposed weights
    for (int idx = tid; idx < 9 * 16 * 64; idx += 256) {
      int t   = idx >> 10;
      int rem = idx & 1023;
      int nn  = rem >> 6;
      int cl  = rem & 63;
      int c   = c0 + cl;
      float f = (c < Cin) ? w[(size_t)(oc0 + nn) * K + c * 9 + t] : 0.f;
      wst[(t * 16 + nn) * 72 + cl] = (__bf16)f;
    }
    __syncthreads();
#pragma unroll
    for (int t = 0; t < 9; ++t) {
      const int row = (n0 + lm + (t - 4) * dil) & 127;
      const __bf16* ab = slabT + row * 72;
      const __bf16* bb = wst + (t * 16 + lm) * 72;
#pragma unroll
      for (int s = 0; s < 2; ++s) {
        const int kl = s << 5;
        FragU A, B;
        const __bf16* ap = ab + kl + (half << 3);
        A.q[0] = LD128(ap);
        A.q[1] = LD128(ap + 16);
        const __bf16* bp = bb + kl + (half << 3);
        B.q[0] = LD128(bp);
        B.q[1] = LD128(bp + 16);
        acc = WMMA_BF16(A.v, B.v, acc);
      }
    }
  }
  const int oc = oc0 + lm;
  const float bs = bias[oc];
  float* ob = out + ((size_t)p * Cout + oc) * N + n0 + (half << 3);
  *reinterpret_cast<float4*>(ob)     = make_float4(acc[0] + bs, acc[1] + bs, acc[2] + bs, acc[3] + bs);
  *reinterpret_cast<float4*>(ob + 4) = make_float4(acc[4] + bs, acc[5] + bs, acc[6] + bs, acc[7] + bs);
}

// ---------------------------------------------------------------------------
// Pointwise GEMM (einsum 'oc,pcn->pon'): out[p,o,n] = sum_c W[o,c]*in[p,c,n]+b[o]
// K is a multiple of 64 for all uses (1024, 1280, 256).
// ---------------------------------------------------------------------------
__global__ __launch_bounds__(256) void k_gemm_pw_wmma(
    const float* __restrict__ in, const float* __restrict__ w, const float* __restrict__ bias,
    float* __restrict__ out, int P, int Cin, int Cout, int N, int relu) {
  extern __shared__ char smem[];
  const int K  = Cin;
  const int Kw = K + 8;
  __bf16* wchT  = (__bf16*)smem;                  // [16n][Kw]
  __bf16* slabT = (__bf16*)(smem + 16 * Kw * 2);  // [128n][72c] (64 used)
  const int tid = threadIdx.x;
  const int oc0 = blockIdx.y << 4;
  for (int idx = tid; idx < (K << 4); idx += 256) {
    int nn = idx / K, k = idx - nn * K;
    wchT[nn * Kw + k] = (__bf16)w[(size_t)(oc0 + nn) * K + k];
  }
  const int wave = tid >> 5, lane = tid & 31;
  const int half = lane >> 4, lm = lane & 15;
  const int p  = blockIdx.x;
  const int n0 = wave << 4;
  const float* ip = in + (size_t)p * Cin * N;
  const __bf16* wrow = wchT + lm * Kw;
  const __bf16* arow = slabT + (n0 + lm) * 72;
  v8f acc = {0.f, 0.f, 0.f, 0.f, 0.f, 0.f, 0.f, 0.f};
  for (int c0 = 0; c0 < K; c0 += 64) {
    __syncthreads();
    for (int idx = tid; idx < 32 * 128; idx += 256) {
      int cpair = idx >> 7;
      int n     = idx & 127;
      int c     = c0 + (cpair << 1);
      v2bf pk;
      pk.x = (__bf16)ip[(size_t)c * N + n];
      pk.y = (__bf16)ip[(size_t)(c + 1) * N + n];
      *reinterpret_cast<v2bf*>(&slabT[n * 72 + (cpair << 1)]) = pk;
    }
    __syncthreads();
#pragma unroll
    for (int s = 0; s < 2; ++s) {
      const int kl = s << 5;
      FragU A, B;
      const __bf16* ap = arow + kl + (half << 3);
      A.q[0] = LD128(ap);
      A.q[1] = LD128(ap + 16);
      const __bf16* bp = wrow + c0 + kl + (half << 3);
      B.q[0] = LD128(bp);
      B.q[1] = LD128(bp + 16);
      acc = WMMA_BF16(A.v, B.v, acc);
    }
  }
  const int oc = oc0 + lm;
  const float bs = bias[oc];
  float4 lo = make_float4(acc[0] + bs, acc[1] + bs, acc[2] + bs, acc[3] + bs);
  float4 hi = make_float4(acc[4] + bs, acc[5] + bs, acc[6] + bs, acc[7] + bs);
  if (relu) {
    lo.x = fmaxf(lo.x, 0.f); lo.y = fmaxf(lo.y, 0.f); lo.z = fmaxf(lo.z, 0.f); lo.w = fmaxf(lo.w, 0.f);
    hi.x = fmaxf(hi.x, 0.f); hi.y = fmaxf(hi.y, 0.f); hi.z = fmaxf(hi.z, 0.f); hi.w = fmaxf(hi.w, 0.f);
  }
  float* ob = out + ((size_t)p * Cout + oc) * N + n0 + (half << 3);
  *reinterpret_cast<float4*>(ob)     = lo;
  *reinterpret_cast<float4*>(ob + 4) = hi;
}

// ---------------------------------------------------------------------------
// GroupNorm: stats per (b, group), then apply (+ optional relu, channel offset)
// ---------------------------------------------------------------------------
__global__ __launch_bounds__(256) void k_gn_stats(
    const float* __restrict__ x, float* __restrict__ mean, float* __restrict__ rstd,
    int C, int HW, int groups, float eps) {
  int bg = blockIdx.x;
  int g = bg % groups, b = bg / groups;
  int cpg = C / groups;
  const float* p = x + ((size_t)b * C + (size_t)g * cpg) * HW;
  size_t n = (size_t)cpg * HW;
  float s = 0.f, s2 = 0.f;
  for (size_t i = threadIdx.x; i < n; i += 256) { float v = p[i]; s += v; s2 += v * v; }
  __shared__ float sh[256], sh2[256];
  sh[threadIdx.x] = s; sh2[threadIdx.x] = s2;
  __syncthreads();
  for (int st = 128; st > 0; st >>= 1) {
    if ((int)threadIdx.x < st) { sh[threadIdx.x] += sh[threadIdx.x + st]; sh2[threadIdx.x] += sh2[threadIdx.x + st]; }
    __syncthreads();
  }
  if (threadIdx.x == 0) {
    float m = sh[0] / (float)n;
    float v = sh2[0] / (float)n - m * m;
    mean[bg] = m;
    rstd[bg] = rsqrtf(fmaxf(v, 0.f) + eps);
  }
}

__global__ __launch_bounds__(256) void k_gn_apply(
    const float* __restrict__ x, const float* __restrict__ gam, const float* __restrict__ bet,
    const float* __restrict__ mean, const float* __restrict__ rstd, float* y,
    int Bn, int C, int HW, int groups, int yC, int yc0, int relu) {
  int i = blockIdx.x * 256 + threadIdx.x;
  int total = Bn * C * HW;
  if (i >= total) return;
  int hw = i % HW;
  int c  = (i / HW) % C;
  int b  = i / (HW * C);
  int bg = b * groups + c / (C / groups);
  float v = (x[i] - mean[bg]) * rstd[bg] * gam[c] + bet[c];
  if (relu) v = fmaxf(v, 0.f);
  y[((size_t)b * yC + yc0 + c) * HW + hw] = v;
}

// ---------------------------------------------------------------------------
// BatchNorm1d (training-mode) of relu(t): stats per channel over (P,N), apply
// with optional residual; also scatter result into the running state tensor.
// ---------------------------------------------------------------------------
__global__ __launch_bounds__(256) void k_bn_stats_relu(
    const float* __restrict__ x, float* __restrict__ mean, float* __restrict__ rstd,
    int P, int C, int N, float eps) {
  int c = blockIdx.x;
  int total = P * N;
  float s = 0.f, s2 = 0.f;
  for (int i = threadIdx.x; i < total; i += 256) {
    int p = i / N, n = i % N;
    float v = fmaxf(x[((size_t)p * C + c) * N + n], 0.f);
    s += v; s2 += v * v;
  }
  __shared__ float sh[256], sh2[256];
  sh[threadIdx.x] = s; sh2[threadIdx.x] = s2;
  __syncthreads();
  for (int st = 128; st > 0; st >>= 1) {
    if ((int)threadIdx.x < st) { sh[threadIdx.x] += sh[threadIdx.x + st]; sh2[threadIdx.x] += sh2[threadIdx.x + st]; }
    __syncthreads();
  }
  if (threadIdx.x == 0) {
    float m = sh[0] / (float)total;
    float v = sh2[0] / (float)total - m * m;
    mean[c] = m;
    rstd[c] = rsqrtf(fmaxf(v, 0.f) + eps);
  }
}

__global__ __launch_bounds__(256) void k_bn_apply(
    const float* __restrict__ t, const float* resid,
    const float* __restrict__ gam, const float* __restrict__ bet,
    const float* __restrict__ mean, const float* __restrict__ rstd,
    float* xout, float* state, int c0, int stateC, int P, int C, int N) {
  int i = blockIdx.x * 256 + threadIdx.x;
  int total = P * C * N;
  if (i >= total) return;
  int n = i % N;
  int c = (i / N) % C;
  int p = i / (N * C);
  float v = fmaxf(t[i], 0.f);
  v = (v - mean[c]) * rstd[c] * gam[c] + bet[c];
  if (resid) v += resid[i];
  xout[i] = v;
  state[((size_t)p * stateC + c0 + c) * N + n] = v;
}

// ---------------------------------------------------------------------------
// Small direct kernels
// ---------------------------------------------------------------------------
__global__ __launch_bounds__(256) void k_conv1x1_sigmoid(
    const float* __restrict__ x, const float* __restrict__ w, const float* __restrict__ bias,
    float* __restrict__ pe, int Bn, int C, int HW) {
  int i = blockIdx.x * 256 + threadIdx.x;
  if (i >= Bn * HW) return;
  int hw = i % HW, b = i / HW;
  float s = bias[0];
  for (int c = 0; c < C; ++c) s += w[c] * x[((size_t)b * C + c) * HW + hw];
  pe[i] = sigf(s);
}

__global__ __launch_bounds__(256) void k_one_minus(const float* __restrict__ a, float* __restrict__ o, int n) {
  int i = blockIdx.x * 256 + threadIdx.x;
  if (i < n) o[i] = 1.f - a[i];
}

__global__ __launch_bounds__(256) void k_att3_sigmoid(
    const float* __restrict__ a, const float* __restrict__ w, const float* __restrict__ bias,
    float* __restrict__ feats, int Bn, int Cin, int H, int W, int FC) {
  int i = blockIdx.x * 256 + threadIdx.x;
  if (i >= Bn * H * W) return;
  int xw = i % W, yh = (i / W) % H, b = i / (W * H);
  float s = bias[0];
  for (int c = 0; c < Cin; ++c)
    for (int ky = 0; ky < 3; ++ky) {
      int iy = yh + ky - 1;
      if (iy < 0 || iy >= H) continue;
      for (int kx = 0; kx < 3; ++kx) {
        int ix = xw + kx - 1;
        if (ix < 0 || ix >= W) continue;
        s += w[(c * 3 + ky) * 3 + kx] * a[((size_t)(b * Cin + c) * H + iy) * W + ix];
      }
    }
  feats[((size_t)b * FC) * H * W + yh * W + xw] = sigf(s);
}

// 4x bilinear upsample (half-pixel centers, clamped edges) -> out part 1
__global__ __launch_bounds__(256) void k_upsample4(
    const float* __restrict__ pe, float* __restrict__ out, int Bn, int Hin, int Win) {
  int Ho = Hin * 4, Wo = Win * 4;
  int i = blockIdx.x * 256 + threadIdx.x;
  if (i >= Bn * Ho * Wo) return;
  int xo = i % Wo, yo = (i / Wo) % Ho, b = i / (Wo * Ho);
  float sx = (xo + 0.5f) * 0.25f - 0.5f;
  float sy = (yo + 0.5f) * 0.25f - 0.5f;
  float fx = floorf(sx), fy = floorf(sy);
  int x0 = (int)fx, y0 = (int)fy;
  float wx = sx - fx, wy = sy - fy;
  int x0c = min(max(x0, 0), Win - 1), x1c = min(max(x0 + 1, 0), Win - 1);
  int y0c = min(max(y0, 0), Hin - 1), y1c = min(max(y0 + 1, 0), Hin - 1);
  const float* pb = pe + (size_t)b * Hin * Win;
  out[i] = (1.f - wx) * (1.f - wy) * pb[y0c * Win + x0c] + wx * (1.f - wy) * pb[y0c * Win + x1c] +
           (1.f - wx) * wy * pb[y1c * Win + x0c] + wx * wy * pb[y1c * Win + x1c];
}

// per-polygon min/max of loc*RO (for de_location)
__global__ __launch_bounds__(128) void k_loc_minmax(const float* __restrict__ loc, float* __restrict__ mnmx, int N) {
  int p = blockIdx.x, n = threadIdx.x;
  float lx = loc[((size_t)p * N + n) * 2 + 0] * 4.f;
  float ly = loc[((size_t)p * N + n) * 2 + 1] * 4.f;
  __shared__ float mnx[128], mny[128], mxx[128], mxy[128];
  mnx[n] = lx; mny[n] = ly; mxx[n] = lx; mxy[n] = ly;
  __syncthreads();
  for (int st = 64; st > 0; st >>= 1) {
    if (n < st) {
      mnx[n] = fminf(mnx[n], mnx[n + st]); mny[n] = fminf(mny[n], mny[n + st]);
      mxx[n] = fmaxf(mxx[n], mxx[n + st]); mxy[n] = fmaxf(mxy[n], mxy[n + st]);
    }
    __syncthreads();
  }
  if (n == 0) { mnmx[p * 4] = mnx[0]; mnmx[p * 4 + 1] = mny[0]; mnmx[p * 4 + 2] = mxx[0]; mnmx[p * 4 + 3] = mxy[0]; }
}

// bilinear sample of 129-ch feature map at contour points; build snake input
// inp (P,130,N): ch 0..127 = features ch1..128; ch 128/129 = de_location(loc*RO)
__global__ __launch_bounds__(256) void k_sample_build(
    const float* __restrict__ fmap, const float* __restrict__ loc, const int* __restrict__ pind,
    const float* __restrict__ mnmx, float* __restrict__ inp, float* __restrict__ attb,
    int P, int N, int FC, int Hf, int Wf) {
  int i = blockIdx.x * 256 + threadIdx.x;
  if (i >= P * N) return;
  int p = i / N, n = i % N;
  float lx = loc[(size_t)i * 2 + 0], ly = loc[(size_t)i * 2 + 1];
  float ix = lx - 0.5f, iy = ly - 0.5f;
  float fx = floorf(ix), fy = floorf(iy);
  int x0 = (int)fx, y0 = (int)fy;
  float wx = ix - fx, wy = iy - fy;
  int b = pind[p];
  float vx0 = (x0 >= 0 && x0 < Wf) ? 1.f : 0.f;
  float vx1 = (x0 + 1 >= 0 && x0 + 1 < Wf) ? 1.f : 0.f;
  float vy0 = (y0 >= 0 && y0 < Hf) ? 1.f : 0.f;
  float vy1 = (y0 + 1 >= 0 && y0 + 1 < Hf) ? 1.f : 0.f;
  int cx0 = min(max(x0, 0), Wf - 1), cx1 = min(max(x0 + 1, 0), Wf - 1);
  int cy0 = min(max(y0, 0), Hf - 1), cy1 = min(max(y0 + 1, 0), Hf - 1);
  float w00 = (1.f - wx) * (1.f - wy) * vx0 * vy0;
  float w10 = wx * (1.f - wy) * vx1 * vy0;
  float w01 = (1.f - wx) * wy * vx0 * vy1;
  float w11 = wx * wy * vx1 * vy1;
  size_t base = (size_t)b * FC * Hf * Wf;
  for (int c = 0; c < FC; ++c) {
    size_t co = base + (size_t)c * Hf * Wf;
    float v = w00 * fmap[co + cy0 * Wf + cx0] + w10 * fmap[co + cy0 * Wf + cx1] +
              w01 * fmap[co + cy1 * Wf + cx0] + w11 * fmap[co + cy1 * Wf + cx1];
    if (c == 0) attb[i] = v;
    else        inp[((size_t)p * 130 + (c - 1)) * N + n] = v;
  }
  float mnx = mnmx[p * 4], mny = mnmx[p * 4 + 1], mxx = mnmx[p * 4 + 2], mxy = mnmx[p * 4 + 3];
  inp[((size_t)p * 130 + 128) * N + n] = (lx * 4.f - mnx) / (mxx - mnx);
  inp[((size_t)p * 130 + 129) * N + n] = (ly * 4.f - mny) / (mxy - mny);
}

// glob = max over n of fus
__global__ __launch_bounds__(256) void k_row_max(const float* __restrict__ fus, float* __restrict__ gmax,
                                                 int P, int C, int N) {
  int i = blockIdx.x * 256 + threadIdx.x;
  if (i >= P * C) return;
  const float* r = fus + (size_t)i * N;
  float m = r[0];
  for (int n = 1; n < N; ++n) m = fmaxf(m, r[n]);
  gmax[i] = m;
}

// state2 (P,1280,N) = concat(glob broadcast(256), state(1024))
__global__ __launch_bounds__(256) void k_build_state2(
    const float* __restrict__ gmax, const float* __restrict__ state, float* __restrict__ st2, int P, int N) {
  size_t i = (size_t)blockIdx.x * 256 + threadIdx.x;
  size_t total = (size_t)P * 1280 * N;
  if (i >= total) return;
  int n = (int)(i % N);
  int c = (int)((i / N) % 1280);
  int p = (int)(i / ((size_t)N * 1280));
  st2[i] = (c < 256) ? gmax[p * 256 + c] : state[((size_t)p * 1024 + (c - 256)) * N + n];
}

// pr3 (2x64) + loc update: newloc = loc*RO + off*att ; locnext = newloc/RO
__global__ __launch_bounds__(256) void k_pr3_update(
    const float* __restrict__ h2, const float* __restrict__ w, const float* __restrict__ bias,
    const float* __restrict__ attb, const float* locin, float* __restrict__ outpred,
    float* locnext, int P, int N) {
  int i = blockIdx.x * 256 + threadIdx.x;
  if (i >= P * N) return;
  int p = i / N, n = i % N;
  float ox = bias[0], oy = bias[1];
  for (int c = 0; c < 64; ++c) {
    float hv = h2[((size_t)p * 64 + c) * N + n];
    ox += w[c] * hv;
    oy += w[64 + c] * hv;
  }
  float a = attb[i];
  float nx = locin[(size_t)i * 2 + 0] * 4.f + ox * a;
  float ny = locin[(size_t)i * 2 + 1] * 4.f + oy * a;
  outpred[(size_t)i * 2 + 0] = nx;
  outpred[(size_t)i * 2 + 1] = ny;
  locnext[(size_t)i * 2 + 0] = nx * 0.25f;
  locnext[(size_t)i * 2 + 1] = ny * 0.25f;
}

// ---------------------------------------------------------------------------
// Host orchestration
// ---------------------------------------------------------------------------
extern "C" void kernel_launch(void* const* d_in, const int* in_sizes, int n_in,
                              void* d_out, int out_size, void* d_ws, size_t ws_size,
                              hipStream_t stream) {
  (void)in_sizes; (void)n_in; (void)out_size; (void)ws_size;
  const float* X    = (const float*)d_in[0];
  const float* LOC0 = (const float*)d_in[1];
  const int*   PIND = (const int*)d_in[2];
  auto PRM = [&](int i) { return (const float*)d_in[i]; };

  const int Bn = 4, H = 256, Wd = 256, HW = H * Wd;
  const int P = 256, N = 128;

  float* ws = (float*)d_ws;
  size_t off = 0;
  auto alloc = [&](size_t n) { float* p = ws + off; off += n; return p; };

  // persistent buffers
  float* pe    = alloc((size_t)Bn * HW);
  float* om    = alloc((size_t)Bn * HW);
  float* feats = alloc((size_t)Bn * 129 * HW);
  float* gmean = alloc(128);
  float* grstd = alloc(128);
  float* bnm   = alloc(128);
  float* bnr   = alloc(128);
  float* mnmx  = alloc(P * 4);
  float* attb  = alloc((size_t)P * N);
  float* locb  = alloc((size_t)P * N * 2);
  float* gmax  = alloc((size_t)P * 256);
  size_t arena = off;
  // CNN temporaries (arena A)
  float* t64   = alloc((size_t)Bn * 64 * HW);
  float* t32a  = alloc((size_t)Bn * 32 * HW);
  float* t32b  = alloc((size_t)Bn * 32 * HW);
  float* t128a = alloc((size_t)Bn * 128 * HW);
  float* t128b = alloc((size_t)Bn * 128 * HW);
  // snake temporaries alias arena A (CNN temps dead once `feats` is built)
  off = arena;
  float* inp   = alloc((size_t)P * 130 * N);
  float* tcv   = alloc((size_t)P * 128 * N);
  float* xbuf  = alloc((size_t)P * 128 * N);
  float* state = alloc((size_t)P * 1024 * N);
  float* fus   = alloc((size_t)P * 256 * N);
  float* st2   = alloc((size_t)P * 1280 * N);
  float* h1    = alloc((size_t)P * 256 * N);
  float* h2    = alloc((size_t)P * 64 * N);

  float* OUT = (float*)d_out;
  const float EPS = 1e-5f;
  dim3 blk(256);
  const unsigned LDS_C3 = 390 * 40 * 2 + 9 * 16 * 40 * 2;   // 42720 B
  const unsigned LDS_CC = 128 * 72 * 2 + 9 * 16 * 72 * 2;   // 39168 B
  auto LDS_PW = [](int K) { return (unsigned)(16 * (K + 8) * 2 + 128 * 72 * 2); };

  // ---- edge branch ----
  k_conv3x3_wmma<<<dim3(2048, 4), blk, LDS_C3, stream>>>(X, PRM(3), t64, Bn, 64, H, Wd, 64, 64, 0);
  k_gn_stats<<<Bn * 32, blk, 0, stream>>>(t64, gmean, grstd, 64, HW, 32, EPS);
  k_gn_apply<<<(Bn * 64 * HW) / 256, blk, 0, stream>>>(t64, PRM(4), PRM(5), gmean, grstd, t64, Bn, 64, HW, 32, 64, 0, 1);
  k_conv1x1_sigmoid<<<(Bn * HW) / 256, blk, 0, stream>>>(t64, PRM(6), PRM(7), pe, Bn, 64, HW);
  k_upsample4<<<(Bn * HW * 16) / 256, blk, 0, stream>>>(pe, OUT, Bn, H, Wd);

  // ---- attention branch ----
  k_one_minus<<<(Bn * HW) / 256, blk, 0, stream>>>(pe, om, Bn * HW);
  k_conv3x3_wmma<<<dim3(2048, 2), blk, LDS_C3, stream>>>(om, PRM(8), t32a, Bn, 1, H, Wd, 32, 32, 0);
  k_gn_stats<<<Bn * 4, blk, 0, stream>>>(t32a, gmean, grstd, 32, HW, 4, EPS);
  k_gn_apply<<<(Bn * 32 * HW) / 256, blk, 0, stream>>>(t32a, PRM(9), PRM(10), gmean, grstd, t32a, Bn, 32, HW, 4, 32, 0, 1);
  k_conv3x3_wmma<<<dim3(2048, 2), blk, LDS_C3, stream>>>(t32a, PRM(11), t32b, Bn, 32, H, Wd, 32, 32, 0);
  k_gn_stats<<<Bn * 4, blk, 0, stream>>>(t32b, gmean, grstd, 32, HW, 4, EPS);
  k_gn_apply<<<(Bn * 32 * HW) / 256, blk, 0, stream>>>(t32b, PRM(12), PRM(13), gmean, grstd, t32b, Bn, 32, HW, 4, 32, 0, 1);
  k_att3_sigmoid<<<(Bn * HW) / 256, blk, 0, stream>>>(t32b, PRM(14), PRM(15), feats, Bn, 32, H, Wd, 129);

  // ---- bottleneck branch (writes features channels 1..128) ----
  k_conv3x3_wmma<<<dim3(2048, 8), blk, LDS_C3, stream>>>(X, PRM(16), t128a, Bn, 64, H, Wd, 128, 128, 0);
  k_gn_stats<<<Bn * 32, blk, 0, stream>>>(t128a, gmean, grstd, 128, HW, 32, EPS);
  k_gn_apply<<<(Bn * 128 * HW) / 256, blk, 0, stream>>>(t128a, PRM(17), PRM(18), gmean, grstd, t128a, Bn, 128, HW, 32, 128, 0, 1);
  k_conv3x3_wmma<<<dim3(2048, 8), blk, LDS_C3, stream>>>(t128a, PRM(19), t128b, Bn, 128, H, Wd, 128, 128, 0);
  k_gn_stats<<<Bn * 32, blk, 0, stream>>>(t128b, gmean, grstd, 128, HW, 32, EPS);
  k_gn_apply<<<(Bn * 128 * HW) / 256, blk, 0, stream>>>(t128b, PRM(20), PRM(21), gmean, grstd, feats, Bn, 128, HW, 32, 129, 1, 1);

  // ---- snake evolution, 3 deformers ----
  const int DILh[7] = {1, 1, 1, 2, 2, 4, 4};
  for (int d = 0; d < 3; ++d) {
    int pb = 22 + d * 16;
    const float* locin = (d == 0) ? LOC0 : locb;
    k_loc_minmax<<<P, 128, 0, stream>>>(locin, mnmx, N);
    k_sample_build<<<(P * N) / 256, blk, 0, stream>>>(feats, locin, PIND, mnmx, inp, attb, P, N, 129, H, Wd);
    // head conv (130 -> 128)
    k_circconv_wmma<<<dim3(P, 8), blk, LDS_CC, stream>>>(inp, PRM(pb + 0), PRM(pb + 1), tcv, P, 130, 128, N, 1);
    k_bn_stats_relu<<<128, blk, 0, stream>>>(tcv, bnm, bnr, P, 128, N, EPS);
    k_bn_apply<<<(P * 128 * N) / 256, blk, 0, stream>>>(tcv, (const float*)nullptr, PRM(pb + 2), PRM(pb + 3),
                                                        bnm, bnr, xbuf, state, 0, 1024, P, 128, N);
    // 7 residual dilated blocks
    for (int i = 0; i < 7; ++i) {
      k_circconv_wmma<<<dim3(P, 8), blk, LDS_CC, stream>>>(
          xbuf, PRM(pb + 4) + (size_t)i * 128 * 128 * 9, PRM(pb + 5) + i * 128, tcv, P, 128, 128, N, DILh[i]);
      k_bn_stats_relu<<<128, blk, 0, stream>>>(tcv, bnm, bnr, P, 128, N, EPS);
      k_bn_apply<<<(P * 128 * N) / 256, blk, 0, stream>>>(tcv, xbuf, PRM(pb + 6) + i * 128, PRM(pb + 7) + i * 128,
                                                          bnm, bnr, xbuf, state, (i + 1) * 128, 1024, P, 128, N);
    }
    // fusion + global max + prediction head
    k_gemm_pw_wmma<<<dim3(P, 16), blk, LDS_PW(1024), stream>>>(state, PRM(pb + 8), PRM(pb + 9), fus, P, 1024, 256, N, 0);
    k_row_max<<<(P * 256) / 256, blk, 0, stream>>>(fus, gmax, P, 256, N);
    k_build_state2<<<(unsigned)(((size_t)P * 1280 * N) / 256), blk, 0, stream>>>(gmax, state, st2, P, N);
    k_gemm_pw_wmma<<<dim3(P, 16), blk, LDS_PW(1280), stream>>>(st2, PRM(pb + 10), PRM(pb + 11), h1, P, 1280, 256, N, 1);
    k_gemm_pw_wmma<<<dim3(P, 4), blk, LDS_PW(256), stream>>>(h1, PRM(pb + 12), PRM(pb + 13), h2, P, 256, 64, N, 1);
    k_pr3_update<<<(P * N) / 256, blk, 0, stream>>>(h2, PRM(pb + 14), PRM(pb + 15), attb, locin,
                                                    OUT + 4194304 + (size_t)d * 65536, locb, P, N);
  }
}
